// TMDLayer_23252952940777
// MI455X (gfx1250) — matrix operand order repaired and checked by reading, not compiled, not run on hardware
//
#include <hip/hip_runtime.h>
#include <hip/hip_bf16.h>

#define N 2048
#define D 768
#define L 16
#define EPSV 0.3f
#define KC 256           // K-chunk staged in LDS by the TDM per block

typedef __attribute__((ext_vector_type(8)))  float        v8f;
typedef __attribute__((ext_vector_type(2)))  float        v2f;
typedef __attribute__((ext_vector_type(16))) __bf16       v16bf;
typedef __attribute__((ext_vector_type(8)))  __bf16       v8bf;
typedef __attribute__((ext_vector_type(4)))  unsigned int v4u;
typedef __attribute__((ext_vector_type(8)))  int          v8i;
typedef __attribute__((ext_vector_type(4)))  int          v4i;

__device__ __forceinline__ __bf16 f2bf(float f) {
    unsigned u = __float_as_uint(f);
    unsigned r = u + 0x7FFFu + ((u >> 16) & 1u);           // round-to-nearest-even
    return __builtin_bit_cast(__bf16, (unsigned short)(r >> 16));
}

__device__ __forceinline__ float block_reduce_sum(float v, float* red) {
    int tid = threadIdx.x;
    red[tid] = v; __syncthreads();
    for (int s = 128; s > 0; s >>= 1) {
        if (tid < s) red[tid] += red[tid + s];
        __syncthreads();
    }
    float r = red[0]; __syncthreads();
    return r;
}

// TDM: load a 2D bf16 tile [64 rows x KC cols] from row-major [*, N] global into
// LDS (rows packed contiguously, KC elems each). D# per CDNA5 ISA ch.8.
// This toolchain exposes the 6-arg builtin: (g0, g1, g2, g3, g4, cpol).
__device__ __forceinline__ void tdm_load_tile(const __bf16* gsrc, unsigned lds_off) {
    unsigned long long ga = (unsigned long long)(uintptr_t)gsrc;
    v4u g0 = { 1u,                                          // count=1, user mode
               lds_off,                                     // D#.lds_addr (bytes)
               (unsigned)ga,                                // global_addr[31:0]
               (unsigned)((ga >> 32) & 0x01FFFFFFull) | (2u << 30) }; // [56:32] | type=2
    v8i g1 = { 0x00010000,                                  // data_size=1 (2 bytes)
               (int)(2048u << 16),                          // tensor_dim0[15:0]=2048
               (int)(768u  << 16),                          // tensor_dim1[15:0]=768
               (int)((unsigned)KC << 16),                   // tile_dim0 = KC
               64,                                          // tile_dim1 = 64 rows
               2048,                                        // tensor_dim0_stride[31:0]
               0, 0 };
    v4i gz4 = { 0, 0, 0, 0 };                               // groups 2/3: unused (2D)
    v8i gz8 = { 0, 0, 0, 0, 0, 0, 0, 0 };
    __builtin_amdgcn_tensor_load_to_lds(g0, g1, gz4, gz4, gz8, 0);
}

// ---------------- 1) x_proj = relu(LN(x @ proj_w + proj_b)) and sq[i]=|xp_i|^2 ---
__global__ void k_proj(const float* __restrict__ x, const float* __restrict__ pw,
                       const float* __restrict__ pb, const float* __restrict__ g,
                       const float* __restrict__ b, float* __restrict__ xp,
                       float* __restrict__ sq) {
    __shared__ float wl[D * L];                            // 48 KB
    int tid = threadIdx.x;
    for (int i = tid; i < D * L; i += 256) wl[i] = pw[i];
    __syncthreads();
    int row = blockIdx.x * 16 + (tid >> 4);
    int l   = tid & 15;
    const float* xr = x + (size_t)row * D;
    float acc = pb[l];
    for (int k = 0; k < D; ++k) acc += xr[k] * wl[k * L + l];
    float s1 = acc, s2 = acc * acc;
    #pragma unroll
    for (int m = 8; m > 0; m >>= 1) {
        s1 += __shfl_xor(s1, m, 16);
        s2 += __shfl_xor(s2, m, 16);
    }
    float mu  = s1 * (1.f / 16.f);
    float var = s2 * (1.f / 16.f) - mu * mu;
    float v   = (acc - mu) * rsqrtf(var + 1e-5f) * g[l] + b[l];
    v = v > 0.f ? v : 0.f;
    xp[row * L + l] = v;
    float ss = v * v;
    #pragma unroll
    for (int m = 8; m > 0; m >>= 1) ss += __shfl_xor(ss, m, 16);
    if (l == 0) sq[row] = ss;
}

// ---------------- 2) pi = sigmoid(relu(xp @ w1 + b1) @ w2 + b2) ----------------
__global__ void k_pi(const float* __restrict__ xp, const float* __restrict__ w1,
                     const float* __restrict__ b1, const float* __restrict__ w2,
                     const float* __restrict__ b2, float* __restrict__ pi) {
    __shared__ float xl[L];
    __shared__ float red[256];
    int row = blockIdx.x, tid = threadIdx.x;
    if (tid < L) xl[tid] = xp[row * L + tid];
    __syncthreads();
    float acc = 0.f;
    #pragma unroll
    for (int c = 0; c < 3; ++c) {
        int d = tid + c * 256;
        float h = b1[d];
        #pragma unroll
        for (int l = 0; l < L; ++l) h += xl[l] * w1[l * D + d];
        h = h > 0.f ? h : 0.f;
        acc += h * w2[d];
    }
    float tot = block_reduce_sum(acc, red);
    if (tid == 0) pi[row] = 1.f / (1.f + __expf(-(tot + b2[0])));
}

// ---------------- 3) Gram via f32 WMMA + fused exp -> K_eps --------------------
__global__ void k_gram_keps(const float* __restrict__ xp, const float* __restrict__ sq,
                            float* __restrict__ keps) {
    int wid  = blockIdx.x * 8 + (threadIdx.x >> 5);        // 4096 waves
    int lane = threadIdx.x & 31;
    int mt = wid >> 5, ns = wid & 31;
    int i0 = mt * 16, n0 = ns * 64;
    int rowa = lane & 15;
    int ksel = (lane >> 4) * 2;
    v8f acc[4] = {};
    for (int k0 = 0; k0 < L; k0 += 4) {
        v2f a = *reinterpret_cast<const v2f*>(xp + (i0 + rowa) * L + k0 + ksel);
        #pragma unroll
        for (int t = 0; t < 4; ++t) {
            int n = n0 + t * 16 + (lane & 15);
            v2f b = *reinterpret_cast<const v2f*>(xp + n * L + k0 + ksel);
            acc[t] = __builtin_amdgcn_wmma_f32_16x16x4_f32(
                false, a, false, b, (short)0, acc[t], false, false);
        }
    }
    int rbase = i0 + ((lane >> 4) << 3);
    #pragma unroll
    for (int t = 0; t < 4; ++t) {
        int j = n0 + t * 16 + (lane & 15);
        float sj = sq[j];
        #pragma unroll
        for (int r = 0; r < 8; ++r) {
            int i = rbase + r;
            float s = sq[i] + sj - 2.f * acc[t][r];
            keps[(size_t)i * N + j] = __expf(-(1.f / (4.f * EPSV)) * s);
        }
    }
}

// ---------------- 4) q_tilde row sums ------------------------------------------
__global__ void k_qsum(const float* __restrict__ keps, float* __restrict__ q) {
    __shared__ float red[256];
    int row = blockIdx.x, tid = threadIdx.x;
    float s = 0.f;
    #pragma unroll
    for (int c = 0; c < 8; ++c) s += keps[(size_t)row * N + tid + c * 256];
    float tot = block_reduce_sum(s, red);
    if (tid == 0) q[row] = tot;
}

// ---------------- 5) v = pi / q ------------------------------------------------
__global__ void k_vv(const float* __restrict__ pi, const float* __restrict__ q,
                     float* __restrict__ vv) {
    int j = blockIdx.x * 256 + threadIdx.x;
    vv[j] = pi[j] / q[j];
}

// ---------------- 6) d[i] = sum_j K*v + 1e-5 -----------------------------------
__global__ void k_dsum(const float* __restrict__ keps, const float* __restrict__ vv,
                       float* __restrict__ dd) {
    __shared__ float red[256];
    int row = blockIdx.x, tid = threadIdx.x;
    float s = 0.f;
    #pragma unroll
    for (int c = 0; c < 8; ++c) {
        int j = tid + c * 256;
        s += keps[(size_t)row * N + j] * vv[j];
    }
    float tot = block_reduce_sum(s, red);
    if (tid == 0) dd[row] = tot + 1e-5f;
}

// ---------------- 7) W[i,j] = bf16( (dt/eps) * K*v[j]/d[i] ) -------------------
__global__ void k_wscale(const float* __restrict__ keps, const float* __restrict__ vv,
                         const float* __restrict__ dd, const float* __restrict__ dt,
                         __bf16* __restrict__ Wb) {
    int base = (blockIdx.x * 256 + threadIdx.x) * 4;
    int i = base >> 11;
    float s = dt[0] * (1.f / EPSV) / dd[i];
    #pragma unroll
    for (int t = 0; t < 4; ++t) {
        int j = (base + t) & (N - 1);
        Wb[base + t] = f2bf(keps[base + t] * vv[j] * s);
    }
}

// ---------------- 8) f32 transpose (f_w -> f_wT) -------------------------------
__global__ void k_transpose_f32(const float* __restrict__ in, float* __restrict__ out,
                                int rows, int cols) {
    __shared__ float tile[32][33];
    int ctiles = cols >> 5;
    int bx = blockIdx.x % ctiles, by = blockIdx.x / ctiles;
    int c0 = bx * 32, r0 = by * 32;
    for (int i = threadIdx.y; i < 32; i += 8)
        tile[i][threadIdx.x] = in[(size_t)(r0 + i) * cols + c0 + threadIdx.x];
    __syncthreads();
    for (int i = threadIdx.y; i < 32; i += 8)
        out[(size_t)(c0 + i) * rows + r0 + threadIdx.x] = tile[threadIdx.x][i];
}

// ---------------- 9) target = x @ f_w + f_b  (f32 WMMA 16x16x4) ----------------
__global__ void k_gemm_target(const float* __restrict__ x, const float* __restrict__ fwT,
                              const float* __restrict__ fb, float* __restrict__ target) {
    int wid  = blockIdx.x * 8 + (threadIdx.x >> 5);        // 1536 waves
    int lane = threadIdx.x & 31;
    int mt = wid / 12, ns = wid % 12;
    int i0 = mt * 16, n0 = ns * 64;
    int rowa = lane & 15;
    int ksel = (lane >> 4) * 2;
    v8f acc[4] = {};
    const float* arow = x + (size_t)(i0 + rowa) * D;
    for (int k0 = 0; k0 < D; k0 += 4) {
        v2f a = *reinterpret_cast<const v2f*>(arow + k0 + ksel);
        #pragma unroll
        for (int t = 0; t < 4; ++t) {
            int n = n0 + t * 16 + (lane & 15);
            v2f b = *reinterpret_cast<const v2f*>(fwT + (size_t)n * D + k0 + ksel);
            acc[t] = __builtin_amdgcn_wmma_f32_16x16x4_f32(
                false, a, false, b, (short)0, acc[t], false, false);
        }
    }
    int rbase = i0 + ((lane >> 4) << 3);
    #pragma unroll
    for (int t = 0; t < 4; ++t) {
        int col = n0 + t * 16 + (lane & 15);
        float bias = fb[col];
        #pragma unroll
        for (int r = 0; r < 8; ++r)
            target[(size_t)(rbase + r) * D + col] = acc[t][r] + bias;
    }
}

// ---------------- 10) Tt = bf16(target^T)  [D][N] ------------------------------
__global__ void k_transpose_bf16(const float* __restrict__ in, __bf16* __restrict__ out) {
    __shared__ float tile[32][33];
    int bx = blockIdx.x % (D >> 5), by = blockIdx.x / (D >> 5);
    int c0 = bx * 32, r0 = by * 32;
    for (int i = threadIdx.y; i < 32; i += 8)
        tile[i][threadIdx.x] = in[(size_t)(r0 + i) * D + c0 + threadIdx.x];
    __syncthreads();
    for (int i = threadIdx.y; i < 32; i += 8)
        out[(size_t)(c0 + i) * N + r0 + threadIdx.x] = f2bf(tile[threadIdx.x][i]);
}

// ---------------- 11) ttr = W @ target  (bf16 WMMA, TDM-staged B panel) --------
// Block = 8 waves computing a 128x64 output tile. The shared 64xKC B panel is
// DMA'd into LDS by the Tensor Data Mover (double-buffered, TENSORcnt-synced).
__global__ void __launch_bounds__(256)
k_gemm_ttr(const __bf16* __restrict__ Wb, const __bf16* __restrict__ Tt,
           float* __restrict__ ttr) {
    __shared__ __bf16 bsm[2][64 * KC];                     // 2 x 32 KB
    int tid  = threadIdx.x;
    int lane = tid & 31;
    int mb = blockIdx.x / 12;                              // 0..15 (128-row blocks)
    int ns = blockIdx.x % 12;                              // 0..11 (64-col strips)
    int i0 = mb * 128 + (tid >> 5) * 16;
    int n0 = ns * 64;
    int rowa = lane & 15;
    int asel = (lane >> 4) * 16;                           // A K-half select (contiguous)
    int bsel = (lane >> 4) * 16;                           // B K-half select
    v8f acc[4] = {};
    const __bf16* arow = Wb + (size_t)(i0 + rowa) * N;
    const __bf16* bsrc = Tt + (size_t)n0 * N;

    if (tid < 32)                                          // wave 0 drives the TDM
        tdm_load_tile(bsrc, (unsigned)(uintptr_t)&bsm[0][0]);

    for (int c = 0; c < N / KC; ++c) {
        __syncthreads();                                   // prev buffer fully consumed
        if (tid < 32) {
            if (c < N / KC - 1) {
                tdm_load_tile(bsrc + (size_t)(c + 1) * KC,
                              (unsigned)(uintptr_t)&bsm[(c + 1) & 1][0]);
                __builtin_amdgcn_s_wait_tensorcnt(1);      // chunk c has landed
            } else {
                __builtin_amdgcn_s_wait_tensorcnt(0);
            }
        }
        __syncthreads();                                   // chunk c visible to all waves
        const __bf16* bl = &bsm[c & 1][0];
        int kc0 = c * KC;
        for (int kk = 0; kk < KC; kk += 32) {
            int k0 = kc0 + kk;
            __builtin_prefetch(arow + k0 + 64, 0, 3);      // global_prefetch_b8
            v8bf alo = *reinterpret_cast<const v8bf*>(arow + k0 + asel);
            v8bf ahi = *reinterpret_cast<const v8bf*>(arow + k0 + 8 + asel);
            v16bf a;
            #pragma unroll
            for (int e = 0; e < 8; ++e) { a[e] = alo[e]; a[e + 8] = ahi[e]; }
            #pragma unroll
            for (int t = 0; t < 4; ++t) {
                int nl = t * 16 + (lane & 15);
                v16bf b = *reinterpret_cast<const v16bf*>(bl + nl * KC + kk + bsel);
                acc[t] = __builtin_amdgcn_wmma_f32_16x16x32_bf16(
                    false, a, false, b, (short)0, acc[t], false, false);
            }
        }
    }
    int rbase = i0 + ((lane >> 4) << 3);
    #pragma unroll
    for (int t = 0; t < 4; ++t) {
        int col = n0 + t * 16 + (lane & 15);
        #pragma unroll
        for (int r = 0; r < 8; ++r)
            ttr[(size_t)(rbase + r) * D + col] = acc[t][r];
    }
}

// ---------------- 12) out = LN(0.7x + (0.3-0.6dt)target + 0.6 ttr) -------------
__global__ void k_final(const float* __restrict__ x, const float* __restrict__ target,
                        const float* __restrict__ ttr, const float* __restrict__ dt,
                        const float* __restrict__ g, const float* __restrict__ b,
                        float* __restrict__ out) {
    __shared__ float red[256];
    int row = blockIdx.x, tid = threadIdx.x;
    float dtv = dt[0];
    float ct  = 0.3f - 0.6f * dtv;
    float t2v[3];
    float s1 = 0.f, s2 = 0.f;
    #pragma unroll
    for (int c = 0; c < 3; ++c) {
        int d = tid + c * 256;
        size_t idx = (size_t)row * D + d;
        float t2 = 0.7f * x[idx] + ct * target[idx] + 0.6f * ttr[idx];
        t2v[c] = t2; s1 += t2; s2 += t2 * t2;
    }
    float tot1 = block_reduce_sum(s1, red);
    float tot2 = block_reduce_sum(s2, red);
    float mu  = tot1 * (1.f / (float)D);
    float var = tot2 * (1.f / (float)D) - mu * mu;
    float rs  = rsqrtf(var + 1e-5f);
    #pragma unroll
    for (int c = 0; c < 3; ++c) {
        int d = tid + c * 256;
        out[(size_t)row * D + d] = (t2v[c] - mu) * rs * g[d] + b[d];
    }
}

extern "C" void kernel_launch(void* const* d_in, const int* in_sizes, int n_in,
                              void* d_out, int out_size, void* d_ws, size_t ws_size,
                              hipStream_t stream) {
    const float* x    = (const float*)d_in[0];
    const float* pw   = (const float*)d_in[1];
    const float* pb   = (const float*)d_in[2];
    const float* l1g  = (const float*)d_in[3];
    const float* l1b  = (const float*)d_in[4];
    const float* w1   = (const float*)d_in[5];
    const float* b1   = (const float*)d_in[6];
    const float* w2   = (const float*)d_in[7];
    const float* b2   = (const float*)d_in[8];
    const float* dt   = (const float*)d_in[9];
    const float* fw   = (const float*)d_in[10];
    const float* fb   = (const float*)d_in[11];
    const float* l2g  = (const float*)d_in[12];
    const float* l2b  = (const float*)d_in[13];
    float* out = (float*)d_out;

    char* w = (char*)d_ws;
    float*  keps   = (float*)w;  w += (size_t)N * N * 4;   // 16.8 MB
    __bf16* Wb     = (__bf16*)w; w += (size_t)N * N * 2;   //  8.4 MB
    float*  target = (float*)w;  w += (size_t)N * D * 4;   //  6.3 MB
    __bf16* Tt     = (__bf16*)w; w += (size_t)D * N * 2;   //  3.1 MB
    float*  ttr    = (float*)w;  w += (size_t)N * D * 4;   //  6.3 MB
    float*  fwT    = (float*)w;  w += (size_t)D * D * 4;   //  2.4 MB
    float*  xp     = (float*)w;  w += (size_t)N * L * 4;
    float*  sq     = (float*)w;  w += (size_t)N * 4;
    float*  q      = (float*)w;  w += (size_t)N * 4;
    float*  pi     = (float*)w;  w += (size_t)N * 4;
    float*  vv     = (float*)w;  w += (size_t)N * 4;
    float*  dd     = (float*)w;  w += (size_t)N * 4;

    k_proj<<<N / 16, 256, 0, stream>>>(x, pw, pb, l1g, l1b, xp, sq);
    k_transpose_f32<<<(D / 32) * (D / 32), dim3(32, 8), 0, stream>>>(fw, fwT, D, D);
    k_pi<<<N, 256, 0, stream>>>(xp, w1, b1, w2, b2, pi);
    k_gram_keps<<<(128 * 32) / 8, 256, 0, stream>>>(xp, sq, keps);
    k_qsum<<<N, 256, 0, stream>>>(keps, q);
    k_vv<<<N / 256, 256, 0, stream>>>(pi, q, vv);
    k_dsum<<<N, 256, 0, stream>>>(keps, vv, dd);
    k_wscale<<<(N * N) / (256 * 4), 256, 0, stream>>>(keps, vv, dd, dt, Wb);
    k_gemm_target<<<(128 * 12) / 8, 256, 0, stream>>>(x, fwT, fb, target);
    k_transpose_bf16<<<(N / 32) * (D / 32), dim3(32, 8), 0, stream>>>(target, Tt);
    k_gemm_ttr<<<16 * 12, 256, 0, stream>>>(Wb, Tt, ttr);
    k_final<<<N, 256, 0, stream>>>(x, target, ttr, dt, l2g, l2b, out);
}
